// Graphormer_48352741818639
// MI455X (gfx1250) — compile-verified, matrix operand order checked by reading, not run on hardware
//
#include <hip/hip_runtime.h>
#include <hip/hip_bf16.h>
#include <math.h>

// Graphormer forward for MI455X (gfx1250): all GEMMs via v_wmma_f32_16x16x32_f16,
// all WMMA operands fed by 16-byte contiguous loads (weights staged transposed).
#define B_    16
#define N_    128
#define S_    129
#define HID_  768
#define H_    32
#define FFN_  3072
#define L_    12
#define DHEAD 24
#define SROWS 160   // padded key rows (5 wmma k-steps of 32)
#define QROWS 144   // padded query rows (9 tiles of 16)
#define KCOLS 160   // padded key columns in score/prob buffers

typedef __attribute__((ext_vector_type(16))) _Float16 v16h;
typedef __attribute__((ext_vector_type(8)))  _Float16 v8h;
typedef __attribute__((ext_vector_type(8)))  float    v8f;

// Load 16 halves in WMMA operand order: elements 0..7 = p[0..7] (K=kb..kb+7),
// elements 8..15 = p[16..23] (K=kb+16..kb+23).  Two global_load_b128.
__device__ inline v16h load16(const _Float16* __restrict__ p) {
  v8h lo = *(const v8h*)p;
  v8h hi = *(const v8h*)(p + 16);
  return __builtin_shufflevector(lo, hi, 0, 1, 2, 3, 4, 5, 6, 7,
                                         8, 9, 10, 11, 12, 13, 14, 15);
}

// ---------------------------------------------------------------------------
// Dual-tile WMMA: 16x32 output per wave.  A is MxK row-major (lda).
// B is "k-contiguous": element (kk, n) at Bt + n*bsn + kk (bsn is a 16B multiple).
// ---------------------------------------------------------------------------
__device__ inline void wmma_dual(const _Float16* __restrict__ A, int lda,
                                 const _Float16* __restrict__ Bt, long bsn,
                                 int row0, int col0, int K, int lane,
                                 v8f& acc0, v8f& acc1) {
  int m    = row0 + (lane & 15);
  int n    = lane & 15;
  int koff = (lane >> 4) << 3;            // +8 halves for lanes 16..31
  const _Float16* arow  = A  + (long)m * lda;
  const _Float16* brow0 = Bt + (long)(col0 + n) * bsn;
  const _Float16* brow1 = Bt + (long)(col0 + 16 + n) * bsn;
  for (int k0 = 0; k0 < K; k0 += 32) {
    int kb = k0 + koff;
    __builtin_prefetch((const void*)(arow + kb + 64), 0, 1);  // global_prefetch_b8
    v16h av  = load16(arow  + kb);
    v16h bv0 = load16(brow0 + kb);
    v16h bv1 = load16(brow1 + kb);
    acc0 = __builtin_amdgcn_wmma_f32_16x16x32_f16(false, av, false, bv0,
                                                  (short)0, acc0, false, false);
    acc1 = __builtin_amdgcn_wmma_f32_16x16x32_f16(false, av, false, bv1,
                                                  (short)0, acc1, false, false);
  }
}

// ---------------------------------------------------------------------------
// Generic GEMM  C = A(MxK) * Bt^T + bias, Bt stored N x K (pre-transposed).
// Epilogues:
//  mode 0 (QK HEAD): out f16 remapped to per-head [bh][s][32] padded, *scale
//  mode 1 (RESID):   resid[row,col] += C   (f32 residual update of h)
//  mode 2 (GELU):    out f16 plain row-major, exact GELU
//  mode 3 (V HEAD):  out f16 transposed per-head [bh][d][SROWS] padded
// ---------------------------------------------------------------------------
__global__ void gemm_f16_epi(const _Float16* __restrict__ A,
                             const _Float16* __restrict__ Bt,
                             const float* __restrict__ bias,
                             int M, int N, int K, int mode, float scale,
                             _Float16* __restrict__ outh,
                             float* __restrict__ resid) {
  int wid    = blockIdx.x * (blockDim.x >> 5) + (threadIdx.x >> 5);
  int tilesN = N >> 5;                       // 32-wide col blocks
  int tiles  = (M >> 4) * tilesN;
  if (wid >= tiles) return;                  // wave-uniform guard
  int tm = wid / tilesN, tn = wid % tilesN;
  int lane = threadIdx.x & 31;
  v8f acc0 = {}, acc1 = {};
  wmma_dual(A, K, Bt, (long)K, tm * 16, tn * 32, K, lane, acc0, acc1);
  int nc = lane & 15, mb = (lane >> 4) * 8;
#pragma unroll
  for (int half = 0; half < 2; ++half) {
    v8f acc = half ? acc1 : acc0;
    int col = tn * 32 + half * 16 + nc;
    float bvv = bias ? bias[col] : 0.f;
#pragma unroll
    for (int rr = 0; rr < 8; ++rr) {
      int row = tm * 16 + mb + rr;
      float v = acc[rr] + bvv;
      if (mode == 1) {
        resid[(long)row * N + col] += v;
      } else if (mode == 2) {
        float g = 0.5f * v * (1.f + erff(v * 0.70710678f));
        outh[(long)row * N + col] = (_Float16)g;
      } else if (mode == 0) {
        int b = row / S_, s = row % S_;
        int hh = col / DHEAD, d = col % DHEAD;
        outh[(((long)(b * H_ + hh)) * SROWS + s) * 32 + d] = (_Float16)(v * scale);
      } else { // mode 3: V transposed per head
        int b = row / S_, s = row % S_;
        int hh = col / DHEAD, d = col % DHEAD;
        outh[(((long)(b * H_ + hh)) * 32 + d) * SROWS + s] = (_Float16)v;
      }
    }
  }
}

// ---------------------------------------------------------------------------
// Attention scores: per (b,h) S[q,k] = sum_d Q[q,d]*K[k,d].  d padded 24->32,
// single WMMA k-step, 16x32 key block per wave.  B operand: Kh[key][32].
// ---------------------------------------------------------------------------
__global__ void attn_scores(const _Float16* __restrict__ Qh,
                            const _Float16* __restrict__ Kh,
                            float* __restrict__ Sbuf) {
  int wid  = blockIdx.x * (blockDim.x >> 5) + (threadIdx.x >> 5);
  int lane = threadIdx.x & 31;
  int bh = wid / 45; int r = wid % 45; int qt = r / 5, kt = r % 5;
  const _Float16* Q  = Qh + (long)bh * SROWS * 32;
  const _Float16* Kp = Kh + (long)bh * SROWS * 32;
  v8f acc0 = {}, acc1 = {};
  wmma_dual(Q, 32, Kp, 32L, qt * 16, kt * 32, 32, lane, acc0, acc1);
  float* So = Sbuf + (long)bh * QROWS * KCOLS;
  int nc = lane & 15, mb = (lane >> 4) * 8;
#pragma unroll
  for (int half = 0; half < 2; ++half) {
    v8f acc = half ? acc1 : acc0;
    int k = kt * 32 + half * 16 + nc;
#pragma unroll
    for (int rr = 0; rr < 8; ++rr)
      So[(long)(qt * 16 + mb + rr) * KCOLS + k] = acc[rr];
  }
}

// ---------------------------------------------------------------------------
// Masked softmax over keys (129 valid), adds gab bias, writes f16 probs into
// padded [QROWS x KCOLS] buffer.  One wave per query row.
// ---------------------------------------------------------------------------
__global__ void attn_softmax(const float* __restrict__ Sbuf,
                             const float* __restrict__ gab,
                             _Float16* __restrict__ Ah) {
  int wid  = blockIdx.x * (blockDim.x >> 5) + (threadIdx.x >> 5);
  int lane = threadIdx.x & 31;
  int bh = wid / QROWS; int q = wid % QROWS;
  _Float16* arow = Ah + ((long)bh * QROWS + q) * KCOLS;
  if (q >= S_) {                           // wave-uniform: zero padded query rows
    for (int k = lane; k < KCOLS; k += 32) arow[k] = (_Float16)0.f;
    return;
  }
  const float* srow = Sbuf + ((long)bh * QROWS + q) * KCOLS;
  int b = bh / H_, hh = bh % H_;
  const float* grow = gab + (((long)b * H_ + hh) * S_ + q) * S_;
  float vals[5];
  float mx = -3.4e38f;
#pragma unroll
  for (int i = 0; i < 5; ++i) {
    int k = lane + 32 * i;
    float v = (k < S_) ? (srow[k] + grow[k]) : -3.4e38f;
    vals[i] = v; mx = fmaxf(mx, v);
  }
  for (int off = 16; off; off >>= 1) mx = fmaxf(mx, __shfl_xor(mx, off, 32));
  float sum = 0.f;
#pragma unroll
  for (int i = 0; i < 5; ++i) {
    int k = lane + 32 * i;
    float e = (k < S_) ? __expf(vals[i] - mx) : 0.f;
    vals[i] = e; sum += e;
  }
  for (int off = 16; off; off >>= 1) sum += __shfl_xor(sum, off, 32);
  float inv = 1.f / sum;
#pragma unroll
  for (int i = 0; i < 5; ++i) arow[lane + 32 * i] = (_Float16)(vals[i] * inv);
}

// ---------------------------------------------------------------------------
// O = P * V per (b,h): K-dim = 160 (5 wmma steps), both 16-col halves (d 0..31,
// valid d<24) done by one wave.  B operand: Vt[bh][d][SROWS] (k-contiguous).
// ---------------------------------------------------------------------------
__global__ void attn_out(const _Float16* __restrict__ Ah,
                         const _Float16* __restrict__ Vt,
                         _Float16* __restrict__ Oh) {
  int wid  = blockIdx.x * (blockDim.x >> 5) + (threadIdx.x >> 5);
  int lane = threadIdx.x & 31;
  int bh = wid / 9; int qt = wid % 9;
  const _Float16* Ap = Ah + (long)bh * QROWS * KCOLS;
  const _Float16* Vp = Vt + (long)bh * 32 * SROWS;
  v8f acc0 = {}, acc1 = {};
  wmma_dual(Ap, KCOLS, Vp, (long)SROWS, qt * 16, 0, KCOLS, lane, acc0, acc1);
  int b = bh / H_, hh = bh % H_;
  int nc = lane & 15, mb = (lane >> 4) * 8;
#pragma unroll
  for (int half = 0; half < 2; ++half) {
    v8f acc = half ? acc1 : acc0;
    int d = half * 16 + nc;
#pragma unroll
    for (int rr = 0; rr < 8; ++rr) {
      int q = qt * 16 + mb + rr;
      if (q < S_ && d < DHEAD)
        Oh[((long)(b * S_ + q)) * HID_ + hh * DHEAD + d] = (_Float16)acc[rr];
    }
  }
}

// ---------------------------------------------------------------------------
// LayerNorm: one 256-thread block per row of h (f32 in), f16 out for GEMMs.
// ---------------------------------------------------------------------------
__global__ void layernorm_f16(const float* __restrict__ hin,
                              const float* __restrict__ g,
                              const float* __restrict__ bb,
                              _Float16* __restrict__ out) {
  __shared__ float s1[256], s2[256];
  int row = blockIdx.x;
  const float* x = hin + (long)row * HID_;
  float a = 0.f, q = 0.f;
  for (int c = threadIdx.x; c < HID_; c += 256) { float v = x[c]; a += v; q += v * v; }
  s1[threadIdx.x] = a; s2[threadIdx.x] = q; __syncthreads();
  for (int off = 128; off; off >>= 1) {
    if (threadIdx.x < off) { s1[threadIdx.x] += s1[threadIdx.x + off];
                             s2[threadIdx.x] += s2[threadIdx.x + off]; }
    __syncthreads();
  }
  float mean = s1[0] / (float)HID_;
  float var  = s2[0] / (float)HID_ - mean * mean;
  float rstd = rsqrtf(var + 1e-5f);
  for (int c = threadIdx.x; c < HID_; c += 256)
    out[(long)row * HID_ + c] = (_Float16)((x[c] - mean) * rstd * g[c] + bb[c]);
}

// ---------------------------------------------------------------------------
// Node features: h[b,0,:]=graph_token; h[b,1+n,:]=sum_j atom_emb[x]+deg embs.
// ---------------------------------------------------------------------------
__global__ void node_feat(const int* __restrict__ x, const int* __restrict__ indeg,
                          const int* __restrict__ outdeg,
                          const float* __restrict__ atom_emb,
                          const float* __restrict__ ide, const float* __restrict__ ode,
                          const float* __restrict__ gtok, float* __restrict__ h) {
  long idx = (long)blockIdx.x * 256 + threadIdx.x;
  const long total = (long)B_ * S_ * HID_;
  if (idx >= total) return;
  int col = (int)(idx % HID_);
  long row = idx / HID_;
  int b = (int)(row / S_), s = (int)(row % S_);
  float v;
  if (s == 0) {
    v = gtok[col];
  } else {
    int n = s - 1;
    v = 0.f;
    const int* xr = x + ((long)(b * N_ + n)) * 9;
#pragma unroll
    for (int j = 0; j < 9; ++j) v += atom_emb[(long)xr[j] * HID_ + col];
    v += ide[(long)indeg[b * N_ + n] * HID_ + col];
    v += ode[(long)outdeg[b * N_ + n] * HID_ + col];
  }
  h[idx] = v;
}

// ---------------------------------------------------------------------------
// gab base: 2*attn_bias + spatial bias + graph-token bias.  Layout (b,h,q,k).
// ---------------------------------------------------------------------------
__global__ void gab_base(const float* __restrict__ ab, const int* __restrict__ sp_pos,
                         const float* __restrict__ sp_emb, const float* __restrict__ gt_vd,
                         float* __restrict__ gab) {
  long idx = (long)blockIdx.x * 256 + threadIdx.x;
  const long total = (long)B_ * H_ * S_ * S_;
  if (idx >= total) return;
  int k = (int)(idx % S_);
  long t = idx / S_;
  int q = (int)(t % S_); t /= S_;
  int hh = (int)(t % H_);
  int b  = (int)(t / H_);
  float v = 2.f * ab[((long)b * S_ + q) * S_ + k];
  if (q == 0 || k == 0) v += gt_vd[hh];
  if (q > 0 && k > 0) {
    int spv = sp_pos[((long)b * N_ + (q - 1)) * N_ + (k - 1)];
    v += sp_emb[(long)spv * H_ + hh];
  }
  gab[idx] = v;
}

// ---------------------------------------------------------------------------
// Edge-encoding bias: one wave per (b,n,m).  Lanes build the edge-mean vector
// (5 x 32) in LDS, each lane (=output head) does 5 HxH matvecs, /sp, adds to gab.
// ---------------------------------------------------------------------------
__global__ void ee_bias(const int* __restrict__ edge_input,
                        const int* __restrict__ sp_pos,
                        const float* __restrict__ edge_emb,
                        const float* __restrict__ W,
                        float* __restrict__ gab) {
  __shared__ float lds[8][5 * 32];
  int w = threadIdx.x >> 5, lane = threadIdx.x & 31;
  long pair = (long)blockIdx.x * 8 + w;            // (b*N + n)*N + m
  int b = (int)(pair / (N_ * N_));
  int r = (int)(pair % (N_ * N_));
  int n = r / N_, m = r % N_;
  const int* ei = edge_input + pair * 5 * 3;
#pragma unroll
  for (int dd = 0; dd < 5; ++dd) {
    const int* e3 = ei + dd * 3;
    float s = edge_emb[(long)e3[0] * H_ + lane] +
              edge_emb[(long)e3[1] * H_ + lane] +
              edge_emb[(long)e3[2] * H_ + lane];
    lds[w][dd * 32 + lane] = s * (1.f / 3.f);
  }
  __syncthreads();
  int sp0 = sp_pos[pair];
  int sp = (sp0 == 0) ? 1 : sp0;
  if (sp > 1) sp -= 1;
  if (sp > 5) sp = 5;
  float acc = 0.f;
#pragma unroll 1
  for (int dd = 0; dd < 5; ++dd)
    for (int hin = 0; hin < 32; ++hin)
      acc += lds[w][dd * 32 + hin] * W[((long)(dd * 32 + hin)) * H_ + lane];
  gab[(((long)(b * H_ + lane)) * S_ + (n + 1)) * S_ + (m + 1)] += acc / (float)sp;
}

// ---------------------------------------------------------------------------
// Stage weights f32 -> f16, TRANSPOSED: src is KxN row-major, dst is NxK.
// Makes the WMMA B operand k-contiguous (two b128 loads per tile per k-step).
// ---------------------------------------------------------------------------
__global__ void f32_to_f16T(const float* __restrict__ s, _Float16* __restrict__ d,
                            int K, int N) {
  long i = (long)blockIdx.x * 256 + threadIdx.x;
  long total = (long)K * N;
  if (i >= total) return;
  int n = (int)(i / K), k = (int)(i % K);
  d[i] = (_Float16)s[(long)k * N + n];
}

__global__ void fill_f16(_Float16* __restrict__ d, long n) {
  long i = (long)blockIdx.x * 256 + threadIdx.x;
  if (i < n) d[i] = (_Float16)0.f;
}

// ---------------------------------------------------------------------------
// Final LN (token 0 only) + projection to scalar per batch.
// ---------------------------------------------------------------------------
__global__ void final_proj(const float* __restrict__ h, const float* __restrict__ g,
                           const float* __restrict__ bb, const float* __restrict__ pw,
                           const float* __restrict__ pb, float* __restrict__ out) {
  __shared__ float s1[256], s2[256];
  int b = blockIdx.x;
  const float* x = h + (long)b * S_ * HID_;    // token 0 row
  float a = 0.f, q = 0.f;
  for (int c = threadIdx.x; c < HID_; c += 256) { float v = x[c]; a += v; q += v * v; }
  s1[threadIdx.x] = a; s2[threadIdx.x] = q; __syncthreads();
  for (int off = 128; off; off >>= 1) {
    if (threadIdx.x < off) { s1[threadIdx.x] += s1[threadIdx.x + off];
                             s2[threadIdx.x] += s2[threadIdx.x + off]; }
    __syncthreads();
  }
  float mean = s1[0] / (float)HID_;
  float var  = s2[0] / (float)HID_ - mean * mean;
  float rstd = rsqrtf(var + 1e-5f);
  __syncthreads();
  float dot = 0.f;
  for (int c = threadIdx.x; c < HID_; c += 256)
    dot += ((x[c] - mean) * rstd * g[c] + bb[c]) * pw[c];
  s1[threadIdx.x] = dot; __syncthreads();
  for (int off = 128; off; off >>= 1) {
    if (threadIdx.x < off) s1[threadIdx.x] += s1[threadIdx.x + off];
    __syncthreads();
  }
  if (threadIdx.x == 0) out[b] = s1[0] + pb[0];
}

// ===========================================================================
extern "C" void kernel_launch(void* const* d_in, const int* in_sizes, int n_in,
                              void* d_out, int out_size, void* d_ws, size_t ws_size,
                              hipStream_t stream) {
  (void)in_sizes; (void)n_in; (void)out_size; (void)ws_size;
  // ---- inputs (setup_inputs order) ----
  const float* attn_bias   = (const float*)d_in[0];
  const int*   spatial_pos = (const int*)  d_in[1];
  const int*   xin         = (const int*)  d_in[2];
  const int*   in_degree   = (const int*)  d_in[3];
  const int*   out_degree  = (const int*)  d_in[4];
  const int*   edge_input  = (const int*)  d_in[5];
  const float* atom_emb    = (const float*)d_in[6];
  const float* edge_emb    = (const float*)d_in[7];
  const float* edge_dis    = (const float*)d_in[8];
  const float* spatial_emb = (const float*)d_in[9];
  const float* in_deg_emb  = (const float*)d_in[10];
  const float* out_deg_emb = (const float*)d_in[11];
  const float* graph_token = (const float*)d_in[12];
  const float* gt_vd       = (const float*)d_in[13];
  const float* ln1_g = (const float*)d_in[14];
  const float* ln1_b = (const float*)d_in[15];
  const float* wq = (const float*)d_in[16]; const float* bq = (const float*)d_in[17];
  const float* wk = (const float*)d_in[18]; const float* bk = (const float*)d_in[19];
  const float* wv = (const float*)d_in[20]; const float* bv = (const float*)d_in[21];
  const float* wo = (const float*)d_in[22]; const float* bo = (const float*)d_in[23];
  const float* ln2_g = (const float*)d_in[24];
  const float* ln2_b = (const float*)d_in[25];
  const float* w1 = (const float*)d_in[26]; const float* b1 = (const float*)d_in[27];
  const float* w2 = (const float*)d_in[28]; const float* b2 = (const float*)d_in[29];
  const float* final_g = (const float*)d_in[30];
  const float* final_b = (const float*)d_in[31];
  const float* proj_w  = (const float*)d_in[32];
  const float* proj_b  = (const float*)d_in[33];
  float* outp = (float*)d_out;

  // ---- workspace bump allocator ----
  char* p = (char*)d_ws;
  auto alloc = [&](size_t bytes) -> char* {
    char* r = p; p += (bytes + 255) & ~(size_t)255; return r;
  };
  float*    gab  = (float*)   alloc((size_t)B_ * H_ * S_ * S_ * 4);
  float*    h    = (float*)   alloc((size_t)B_ * S_ * HID_ * 4);
  _Float16* Xh   = (_Float16*)alloc((size_t)B_ * S_ * HID_ * 2);
  _Float16* Qh   = (_Float16*)alloc((size_t)B_ * H_ * SROWS * 32 * 2);
  _Float16* Kh   = (_Float16*)alloc((size_t)B_ * H_ * SROWS * 32 * 2);
  _Float16* Vt   = (_Float16*)alloc((size_t)B_ * H_ * 32 * SROWS * 2);
  float*    Sbuf = (float*)   alloc((size_t)B_ * H_ * QROWS * KCOLS * 4);
  _Float16* Ah   = (_Float16*)alloc((size_t)B_ * H_ * QROWS * KCOLS * 2);
  _Float16* Oh   = (_Float16*)alloc((size_t)B_ * S_ * HID_ * 2);
  _Float16* Fh   = (_Float16*)alloc((size_t)B_ * S_ * FFN_ * 2);
  _Float16* WqT  = (_Float16*)alloc((size_t)HID_ * HID_ * 2);
  _Float16* WkT  = (_Float16*)alloc((size_t)HID_ * HID_ * 2);
  _Float16* WvT  = (_Float16*)alloc((size_t)HID_ * HID_ * 2);
  _Float16* WoT  = (_Float16*)alloc((size_t)HID_ * HID_ * 2);
  _Float16* W1T  = (_Float16*)alloc((size_t)HID_ * FFN_ * 2);
  _Float16* W2T  = (_Float16*)alloc((size_t)FFN_ * HID_ * 2);

  const int TPB = 256;
  auto g1 = [](long n) { return (int)((n + 255) / 256); };       // per-thread grids
  auto gw = [](long waves) { return (int)((waves + 7) / 8); };   // per-wave grids

  const long nPad = (long)B_ * H_ * SROWS * 32;
  // zero padded Q/K/V once per call (valid regions overwritten every layer)
  fill_f16<<<g1(nPad), TPB, 0, stream>>>(Qh, nPad);
  fill_f16<<<g1(nPad), TPB, 0, stream>>>(Kh, nPad);
  fill_f16<<<g1(nPad), TPB, 0, stream>>>(Vt, nPad);

  // node features + attention bias
  node_feat<<<g1((long)B_ * S_ * HID_), TPB, 0, stream>>>(
      xin, in_degree, out_degree, atom_emb, in_deg_emb, out_deg_emb, graph_token, h);
  gab_base<<<g1((long)B_ * H_ * S_ * S_), TPB, 0, stream>>>(
      attn_bias, spatial_pos, spatial_emb, gt_vd, gab);
  ee_bias<<<(B_ * N_ * N_) / 8, TPB, 0, stream>>>(
      edge_input, spatial_pos, edge_emb, edge_dis, gab);

  const int M = B_ * S_;                        // 2064 = 129 row tiles exactly
  const float scale = 0.20412414523193154f;     // 24^-0.5
  const long nHH = (long)HID_ * HID_, nHF = (long)HID_ * FFN_;

  for (int l = 0; l < L_; ++l) {
    // stage this layer's weights as f16, transposed (N x K)
    f32_to_f16T<<<g1(nHH), TPB, 0, stream>>>(wq + (long)l * nHH, WqT, HID_, HID_);
    f32_to_f16T<<<g1(nHH), TPB, 0, stream>>>(wk + (long)l * nHH, WkT, HID_, HID_);
    f32_to_f16T<<<g1(nHH), TPB, 0, stream>>>(wv + (long)l * nHH, WvT, HID_, HID_);
    f32_to_f16T<<<g1(nHH), TPB, 0, stream>>>(wo + (long)l * nHH, WoT, HID_, HID_);
    f32_to_f16T<<<g1(nHF), TPB, 0, stream>>>(w1 + (long)l * nHF, W1T, HID_, FFN_);
    f32_to_f16T<<<g1(nHF), TPB, 0, stream>>>(w2 + (long)l * nHF, W2T, FFN_, HID_);

    // LN1 -> Xh (f16)
    layernorm_f16<<<M, TPB, 0, stream>>>(h, ln1_g + l * HID_, ln1_b + l * HID_, Xh);

    // Q/K/V projections (WMMA), per-head padded layouts
    long qkvTiles = (long)(M / 16) * (HID_ / 32);
    gemm_f16_epi<<<gw(qkvTiles), TPB, 0, stream>>>(Xh, WqT, bq + l * HID_,
        M, HID_, HID_, 0, scale, Qh, nullptr);
    gemm_f16_epi<<<gw(qkvTiles), TPB, 0, stream>>>(Xh, WkT, bk + l * HID_,
        M, HID_, HID_, 0, 1.f, Kh, nullptr);
    gemm_f16_epi<<<gw(qkvTiles), TPB, 0, stream>>>(Xh, WvT, bv + l * HID_,
        M, HID_, HID_, 3, 1.f, Vt, nullptr);

    // scores (WMMA, single k-step), softmax(+gab), P*V (WMMA)
    attn_scores <<<gw((long)B_ * H_ * 9 * 5), TPB, 0, stream>>>(Qh, Kh, Sbuf);
    attn_softmax<<<gw((long)B_ * H_ * QROWS), TPB, 0, stream>>>(Sbuf, gab, Ah);
    attn_out    <<<gw((long)B_ * H_ * 9), TPB, 0, stream>>>(Ah, Vt, Oh);

    // output projection + residual (WMMA)
    gemm_f16_epi<<<gw(qkvTiles), TPB, 0, stream>>>(Oh, WoT, bo + l * HID_,
        M, HID_, HID_, 1, 1.f, nullptr, h);

    // LN2 -> Xh, FFN1 (+GELU), FFN2 (+residual)
    layernorm_f16<<<M, TPB, 0, stream>>>(h, ln2_g + l * HID_, ln2_b + l * HID_, Xh);
    long f1Tiles = (long)(M / 16) * (FFN_ / 32);
    gemm_f16_epi<<<gw(f1Tiles), TPB, 0, stream>>>(Xh, W1T, b1 + (long)l * FFN_,
        M, FFN_, HID_, 2, 1.f, Fh, nullptr);
    gemm_f16_epi<<<gw(qkvTiles), TPB, 0, stream>>>(Fh, W2T, b2 + l * HID_,
        M, HID_, FFN_, 1, 1.f, nullptr, h);
  }

  final_proj<<<B_, TPB, 0, stream>>>(h, final_g, final_b, proj_w, proj_b, outp);
}
